// VolSelfAttention_83803401880145
// MI455X (gfx1250) — matrix operand order repaired and checked by reading, not compiled, not run on hardware
//
#include <hip/hip_runtime.h>

// ---------------------------------------------------------------------------
// MI455X (gfx1250) fused implementation.
// All GEMM work (qkv / qkvC-1x1 / proj / projC, plus both attentions) runs on
// v_wmma_f32_16x16x32_f16 (f16 in, f32 accumulate).  Softmax / cosine-norm /
// depthwise-3x3 / gating are VALU.  The "Gao" gate is a constant 64-vector
// because softmax rows sum to 1 (pooled == 1/64) and 3x3 conv on 1x1 input is
// its center tap -> computed once from weights, fused into the final GEMM.
// ---------------------------------------------------------------------------

typedef _Float16 f16;
typedef __attribute__((ext_vector_type(16))) _Float16 v16h;
typedef __attribute__((ext_vector_type(8)))  float    v8f;
typedef __attribute__((ext_vector_type(4)))  float    f32x4;

union FragU { v16h v; f32x4 q[2]; };

// ---- WMMA wrapper ----------------------------------------------------------
__device__ __forceinline__ v8f wmma_f16(v16h a, v16h b, v8f c) {
  return __builtin_amdgcn_wmma_f32_16x16x32_f16(
      /*neg_a=*/false, a, /*neg_b=*/false, b,
      /*c_mod=*/(short)0, c, /*reuse_a=*/false, /*reuse_b=*/false);
}

// ---- Fragment loaders ------------------------------------------------------
// A (16x32 f16, row-major M x K, `stride` in elements):
//   lane-half kb = 0/16 ; a[j]=K kb+j, j = 0..15
__device__ __forceinline__ v16h load_fragA(const f16* tile, int stride, int lane) {
  int r  = lane & 15;
  int kb = (lane < 16) ? 0 : 16;
  const f16* p = tile + r * stride + kb;
  FragU u;
  u.q[0] = *reinterpret_cast<const f32x4*>(p);
  u.q[1] = *reinterpret_cast<const f32x4*>(p + 8);
  return u.v;
}
// B (32x16 f16) loaded from row-major B^T (N x K): lane-half kb = 0/16,
// b[j] = BT[n][kb + j], j = 0..15 (contiguous 32B per lane)
__device__ __forceinline__ v16h load_fragB_rows(const f16* tileT, int stride, int lane) {
  int n  = lane & 15;
  int kb = (lane < 16) ? 0 : 16;
  const f16* p = tileT + n * stride + kb;
  FragU u;
  u.q[0] = *reinterpret_cast<const f32x4*>(p);
  u.q[1] = *reinterpret_cast<const f32x4*>(p + 8);
  return u.v;
}
// B (32x16) gathered from row-major B (K x N): b[j] = tile[(kb+j)*stride + n]
__device__ __forceinline__ v16h load_fragB_gather(const f16* tile, int stride, int lane) {
  int n  = lane & 15;
  int kb = (lane < 16) ? 0 : 16;
  v16h b;
#pragma unroll
  for (int j = 0; j < 16; ++j) b[j] = tile[(kb + j) * stride + n];
  return b;
}

// ---- Small helpers ---------------------------------------------------------
__global__ void cvt_kernel(const float* __restrict__ src, f16* __restrict__ dst, int n) {
  int i = blockIdx.x * blockDim.x + threadIdx.x;
  int stride = gridDim.x * blockDim.x;
  for (; i < n; i += stride) dst[i] = (f16)src[i];
}

__global__ void bcat_kernel(const float* __restrict__ qkv_b, float* __restrict__ bcat) {
  int i = blockIdx.x * blockDim.x + threadIdx.x;
  if (i < 3072) bcat[i] = (i < 1536) ? qkv_b[i] : 0.0f;
}

// gate[n] = 1 + conv2(BN(conv1(1/64))) using only the 3x3 center taps
__global__ void gate_kernel(const float* __restrict__ gao1_w, const float* __restrict__ gao1_b,
                            const float* __restrict__ bn_g,  const float* __restrict__ bn_b,
                            const float* __restrict__ gao2_w, const float* __restrict__ gao2_b,
                            float* __restrict__ gate) {
  __shared__ float g1[32];
  int tid = threadIdx.x;  // 64 threads
  if (tid < 32) {
    float s = 0.f;
    for (int hh = 0; hh < 16; ++hh) s += gao1_w[((tid * 16 + hh) * 3 + 1) * 3 + 1];
    float v = s * (1.0f / 64.0f) + gao1_b[tid];
    v = v * (bn_g[tid] * rsqrtf(1.0f + 1e-5f)) + bn_b[tid];
    g1[tid] = v;
  }
  __syncthreads();
  float s2 = gao2_b[tid];
  for (int c = 0; c < 32; ++c) s2 += gao2_w[((tid * 32 + c) * 3 + 1) * 3 + 1] * g1[c];
  gate[tid] = 1.0f + s2;
}

// ---- K1: qkv + qkvC 1x1 as one GEMM: [65536,512] x [3072,512]^T ------------
__global__ void __launch_bounds__(32) gemm_qkv_kernel(
    const f16* __restrict__ xh, const f16* __restrict__ wcat,
    const float* __restrict__ bcat, f16* __restrict__ qkv_out) {
  const int lane = threadIdx.x;
  const int nb = blockIdx.x;   // 0..47
  const int mb = blockIdx.y;   // 0..1023
  v8f zero = {};
  v8f acc[4][4];
#pragma unroll
  for (int i = 0; i < 4; ++i)
#pragma unroll
    for (int j = 0; j < 4; ++j) acc[i][j] = zero;

  const f16* Abase = xh   + (size_t)(mb * 64) * 512;
  const f16* Bbase = wcat + (size_t)(nb * 64) * 512;
  for (int ks = 0; ks < 16; ++ks) {
    __builtin_prefetch(Abase + ((ks + 1) & 15) * 32, 0, 0);
    v16h A[4], Bm[4];
#pragma unroll
    for (int mi = 0; mi < 4; ++mi) A[mi]  = load_fragA(Abase + (mi * 16) * 512 + ks * 32, 512, lane);
#pragma unroll
    for (int nj = 0; nj < 4; ++nj) Bm[nj] = load_fragB_rows(Bbase + (nj * 16) * 512 + ks * 32, 512, lane);
#pragma unroll
    for (int mi = 0; mi < 4; ++mi)
#pragma unroll
      for (int nj = 0; nj < 4; ++nj)
        acc[mi][nj] = wmma_f16(A[mi], Bm[nj], acc[mi][nj]);
  }
  __shared__ alignas(16) f16 sC[64 * 64];
  const int half8 = (lane >= 16) ? 8 : 0;
  const int cl = lane & 15;
#pragma unroll
  for (int nj = 0; nj < 4; ++nj) {
    int c = nj * 16 + cl;
    float bias = bcat[nb * 64 + c];
#pragma unroll
    for (int mi = 0; mi < 4; ++mi)
#pragma unroll
      for (int g = 0; g < 8; ++g) {
        int r = mi * 16 + g + half8;
        sC[r * 64 + c] = (f16)(acc[mi][nj][g] + bias);
      }
  }
  __syncthreads();
#pragma unroll
  for (int rr = 0; rr < 2; ++rr) {
    int r = lane * 2 + rr;
    const f32x4* src = reinterpret_cast<const f32x4*>(sC + r * 64);
    f32x4* dst = reinterpret_cast<f32x4*>(qkv_out + (size_t)(mb * 64 + r) * 3072 + nb * 64);
#pragma unroll
    for (int j = 0; j < 8; ++j) dst[j] = src[j];
  }
}

// ---- K2: depthwise 3x3 (pad 1) over 8x8, 1536 channels ---------------------
__global__ void dwconv_kernel(const f16* __restrict__ qkv_out, const float* __restrict__ dwW,
                              f16* __restrict__ dwout) {
  int t = blockIdx.x;                  // 0..65535
  int b = t >> 6, p = t & 63, y = p >> 3, x = p & 7;
  for (int o = threadIdx.x; o < 1536; o += blockDim.x) {
    float s = 0.f;
#pragma unroll
    for (int dy = 0; dy < 3; ++dy) {
      int yy = y + dy - 1; if (yy < 0 || yy > 7) continue;
#pragma unroll
      for (int dx = 0; dx < 3; ++dx) {
        int xx = x + dx - 1; if (xx < 0 || xx > 7) continue;
        s += (float)qkv_out[(size_t)(b * 64 + yy * 8 + xx) * 3072 + 1536 + o] *
             dwW[o * 9 + dy * 3 + dx];
      }
    }
    dwout[(size_t)t * 1536 + o] = (f16)s;
  }
}

// ---- K3: windowed spatial attention, one wave per (b, head) ----------------
__global__ void __launch_bounds__(32) spatial_attn_kernel(
    const f16* __restrict__ qkv_out, const float* __restrict__ rpb_table,
    f16* __restrict__ x1pre) {
  const int lane = threadIdx.x;
  const int bh = blockIdx.x;           // 0..16383
  const int b = bh >> 4, h = bh & 15;
  const int t0 = b * 64;
  const f16* qptr = qkv_out + (size_t)t0 * 3072 + h * 32;
  const f16* kptr = qptr + 512;
  const f16* vptr = qptr + 1024;

  __shared__ alignas(16) f16 sP[64 * 64];
  __shared__ alignas(16) f16 sV[64 * 32];

  // stage V (coalesced): 2 rows per lane
#pragma unroll
  for (int rr = 0; rr < 2; ++rr) {
    int r = lane * 2 + rr;
    const f32x4* src = reinterpret_cast<const f32x4*>(vptr + (size_t)r * 3072);
    f32x4* dst = reinterpret_cast<f32x4*>(sV + r * 32);
#pragma unroll
    for (int j = 0; j < 4; ++j) dst[j] = src[j];
  }

  // S = Q K^T  (64x64, K=32 -> single k-step)
  v16h Aq[4], Bk[4];
#pragma unroll
  for (int mi = 0; mi < 4; ++mi) Aq[mi] = load_fragA(qptr + (size_t)(mi * 16) * 3072, 3072, lane);
#pragma unroll
  for (int nj = 0; nj < 4; ++nj) Bk[nj] = load_fragB_rows(kptr + (size_t)(nj * 16) * 3072, 3072, lane);
  v8f zero = {};
  v8f S[4][4];
#pragma unroll
  for (int mi = 0; mi < 4; ++mi)
#pragma unroll
    for (int nj = 0; nj < 4; ++nj)
      S[mi][nj] = wmma_f16(Aq[mi], Bk[nj], zero);

  // scale + gathered rel-pos bias + row softmax (half-wave shfl reductions)
  const float scale = 0.17677669529663687f;  // 1/sqrt(32)
  const int half8 = (lane >= 16) ? 8 : 0;
  const int cl = lane & 15;
#pragma unroll
  for (int mi = 0; mi < 4; ++mi) {
#pragma unroll
    for (int g = 0; g < 8; ++g) {
      int r = mi * 16 + g + half8;
      int y1 = r >> 3, xx1 = r & 7;
      float v[4];
#pragma unroll
      for (int nj = 0; nj < 4; ++nj) {
        int c = nj * 16 + cl;
        int y2 = c >> 3, xx2 = c & 7;
        int idx = (y1 - y2 + 7) * 15 + (xx1 - xx2 + 7);
        v[nj] = S[mi][nj][g] * scale + rpb_table[idx * 16 + h];
      }
      float mx = fmaxf(fmaxf(v[0], v[1]), fmaxf(v[2], v[3]));
#pragma unroll
      for (int m = 1; m < 16; m <<= 1) mx = fmaxf(mx, __shfl_xor(mx, m, 32));
      float e[4]; float sum = 0.f;
#pragma unroll
      for (int nj = 0; nj < 4; ++nj) { e[nj] = __expf(v[nj] - mx); sum += e[nj]; }
#pragma unroll
      for (int m = 1; m < 16; m <<= 1) sum += __shfl_xor(sum, m, 32);
      float inv = 1.0f / sum;
#pragma unroll
      for (int nj = 0; nj < 4; ++nj) sP[r * 64 + nj * 16 + cl] = (f16)(e[nj] * inv);
    }
  }
  __syncthreads();

  // O = P V  (64x32, K=64 -> 2 k-steps)
  v8f O[4][2];
#pragma unroll
  for (int mi = 0; mi < 4; ++mi)
#pragma unroll
    for (int nt = 0; nt < 2; ++nt) O[mi][nt] = zero;
#pragma unroll
  for (int ks = 0; ks < 2; ++ks) {
    v16h Ap[4], Bv[2];
#pragma unroll
    for (int mi = 0; mi < 4; ++mi) Ap[mi] = load_fragA(sP + (mi * 16) * 64 + ks * 32, 64, lane);
#pragma unroll
    for (int nt = 0; nt < 2; ++nt) Bv[nt] = load_fragB_gather(sV + (ks * 32) * 32 + nt * 16, 32, lane);
#pragma unroll
    for (int mi = 0; mi < 4; ++mi)
#pragma unroll
      for (int nt = 0; nt < 2; ++nt)
        O[mi][nt] = wmma_f16(Ap[mi], Bv[nt], O[mi][nt]);
  }
  // store into x1pre[t][h*32+d]
#pragma unroll
  for (int mi = 0; mi < 4; ++mi)
#pragma unroll
    for (int nt = 0; nt < 2; ++nt)
#pragma unroll
      for (int g = 0; g < 8; ++g) {
        int r = mi * 16 + g + half8;     // query token
        int d = nt * 16 + cl;            // head-dim index
        x1pre[(size_t)(t0 + r) * 512 + h * 32 + d] = (f16)O[mi][nt][g];
      }
}

// ---- K4: cosine channel attention, one wave per (b, head) ------------------
__global__ void __launch_bounds__(32) channel_attn_kernel(
    const f16* __restrict__ dwout, const float* __restrict__ temperature,
    f16* __restrict__ x2pre) {
  const int lane = threadIdx.x;
  const int bh = blockIdx.x;
  const int b = bh >> 4, h = bh & 15;
  const int t0 = b * 64;
  __shared__ alignas(16) f16 sQ[32 * 64];
  __shared__ alignas(16) f16 sK[32 * 64];
  __shared__ alignas(16) f16 sVc[32 * 64];
  __shared__ alignas(16) f16 sP[32 * 32];
  __shared__ float sQs[32], sKs[32];

  // stage transposed: row = channel (lane), col = token
  for (int p = 0; p < 64; ++p) {
    size_t base = (size_t)(t0 + p) * 1536 + h * 32 + lane;
    sQ[lane * 64 + p]  = dwout[base];
    sK[lane * 64 + p]  = dwout[base + 512];
    sVc[lane * 64 + p] = dwout[base + 1024];
  }
  __syncthreads();
  // per-channel L2 norms along tokens
  float sq = 0.f, sk = 0.f;
  for (int p = 0; p < 64; ++p) {
    float q = (float)sQ[lane * 64 + p]; sq += q * q;
    float k = (float)sK[lane * 64 + p]; sk += k * k;
  }
  sQs[lane] = 1.0f / fmaxf(sqrtf(sq), 1e-12f);
  sKs[lane] = 1.0f / fmaxf(sqrtf(sk), 1e-12f);
  __syncthreads();

  // S = Q K^T  (32x32, K=64 -> 2 k-steps); norms folded in post-GEMM
  v8f zero = {};
  v8f S[2][2];
#pragma unroll
  for (int i = 0; i < 2; ++i)
#pragma unroll
    for (int j = 0; j < 2; ++j) S[i][j] = zero;
#pragma unroll
  for (int ks = 0; ks < 2; ++ks) {
    v16h Aq[2], Bk[2];
#pragma unroll
    for (int mi = 0; mi < 2; ++mi) Aq[mi] = load_fragA(sQ + (mi * 16) * 64 + ks * 32, 64, lane);
#pragma unroll
    for (int nj = 0; nj < 2; ++nj) Bk[nj] = load_fragB_rows(sK + (nj * 16) * 64 + ks * 32, 64, lane);
#pragma unroll
    for (int mi = 0; mi < 2; ++mi)
#pragma unroll
      for (int nj = 0; nj < 2; ++nj)
        S[mi][nj] = wmma_f16(Aq[mi], Bk[nj], S[mi][nj]);
  }
  const float temp  = temperature[h];
  const float decay = logf(1.0f - exp2f(-2.0f - 0.25f * (float)h));
  const int half8 = (lane >= 16) ? 8 : 0;
  const int cl = lane & 15;
#pragma unroll
  for (int mi = 0; mi < 2; ++mi) {
#pragma unroll
    for (int g = 0; g < 8; ++g) {
      int r = mi * 16 + g + half8;
      float qs = sQs[r];
      float v[2];
#pragma unroll
      for (int nj = 0; nj < 2; ++nj) {
        int c = nj * 16 + cl;
        v[nj] = S[mi][nj][g] * qs * sKs[c] * temp + fabsf((float)(r - c)) * decay;
      }
      float mx = fmaxf(v[0], v[1]);
#pragma unroll
      for (int m = 1; m < 16; m <<= 1) mx = fmaxf(mx, __shfl_xor(mx, m, 32));
      float e0 = __expf(v[0] - mx), e1 = __expf(v[1] - mx);
      float sum = e0 + e1;
#pragma unroll
      for (int m = 1; m < 16; m <<= 1) sum += __shfl_xor(sum, m, 32);
      float inv = 1.0f / sum;
      sP[r * 32 + cl]      = (f16)(e0 * inv);
      sP[r * 32 + 16 + cl] = (f16)(e1 * inv);
    }
  }
  __syncthreads();

  // O = P Vc  (32x64, K=32 -> single k-step)
  v8f O[2][4];
#pragma unroll
  for (int mi = 0; mi < 2; ++mi)
#pragma unroll
    for (int nt = 0; nt < 4; ++nt) O[mi][nt] = zero;
  v16h Ap[2];
#pragma unroll
  for (int mi = 0; mi < 2; ++mi) Ap[mi] = load_fragA(sP + (mi * 16) * 32, 32, lane);
#pragma unroll
  for (int nt = 0; nt < 4; ++nt) {
    v16h Bv = load_fragB_gather(sVc + nt * 16, 64, lane);
#pragma unroll
    for (int mi = 0; mi < 2; ++mi) O[mi][nt] = wmma_f16(Ap[mi], Bv, O[mi][nt]);
  }
  // store into x2pre[t][h*32+cd]
#pragma unroll
  for (int mi = 0; mi < 2; ++mi)
#pragma unroll
    for (int nt = 0; nt < 4; ++nt)
#pragma unroll
      for (int g = 0; g < 8; ++g) {
        int cd = mi * 16 + g + half8;    // channel
        int p  = nt * 16 + cl;           // token
        x2pre[(size_t)(t0 + p) * 512 + h * 32 + cd] = (f16)O[mi][nt][g];
      }
}

// ---- K5: out = x1pre@Wp^T + pb + gate[n]*(x2pre@Wc^T + cb) -----------------
__global__ void __launch_bounds__(32) final_gemm_kernel(
    const f16* __restrict__ x1pre, const f16* __restrict__ x2pre,
    const f16* __restrict__ wp, const f16* __restrict__ wc,
    const float* __restrict__ proj_b, const float* __restrict__ projC_b,
    const float* __restrict__ gate, float* __restrict__ out) {
  const int lane = threadIdx.x;
  const int nb = blockIdx.x;   // 0..7
  const int mb = blockIdx.y;   // 0..1023
  v8f zero = {};
  v8f acc[4][4];
#pragma unroll
  for (int i = 0; i < 4; ++i)
#pragma unroll
    for (int j = 0; j < 4; ++j) acc[i][j] = zero;

  const int half8 = (lane >= 16) ? 8 : 0;
  const int cl = lane & 15;

  // --- x2 GEMM ---
  {
    const f16* Abase = x2pre + (size_t)(mb * 64) * 512;
    const f16* Bbase = wc    + (size_t)(nb * 64) * 512;
    for (int ks = 0; ks < 16; ++ks) {
      v16h A[4], Bm[4];
#pragma unroll
      for (int mi = 0; mi < 4; ++mi) A[mi]  = load_fragA(Abase + (mi * 16) * 512 + ks * 32, 512, lane);
#pragma unroll
      for (int nj = 0; nj < 4; ++nj) Bm[nj] = load_fragB_rows(Bbase + (nj * 16) * 512 + ks * 32, 512, lane);
#pragma unroll
      for (int mi = 0; mi < 4; ++mi)
#pragma unroll
        for (int nj = 0; nj < 4; ++nj)
          acc[mi][nj] = wmma_f16(A[mi], Bm[nj], acc[mi][nj]);
    }
  }
  // --- epilogue 1: acc = (acc + projC_b) * gate[token] ---
  float cbv[4];
#pragma unroll
  for (int nj = 0; nj < 4; ++nj) cbv[nj] = projC_b[nb * 64 + nj * 16 + cl];
#pragma unroll
  for (int mi = 0; mi < 4; ++mi)
#pragma unroll
    for (int g = 0; g < 8; ++g) {
      int n = mi * 16 + g + half8;       // token-in-window (mb*64 is multiple of 64)
      float gt = gate[n];
#pragma unroll
      for (int nj = 0; nj < 4; ++nj)
        acc[mi][nj][g] = (acc[mi][nj][g] + cbv[nj]) * gt;
    }
  // --- x1 GEMM accumulates into the same C ---
  {
    const f16* Abase = x1pre + (size_t)(mb * 64) * 512;
    const f16* Bbase = wp    + (size_t)(nb * 64) * 512;
    for (int ks = 0; ks < 16; ++ks) {
      v16h A[4], Bm[4];
#pragma unroll
      for (int mi = 0; mi < 4; ++mi) A[mi]  = load_fragA(Abase + (mi * 16) * 512 + ks * 32, 512, lane);
#pragma unroll
      for (int nj = 0; nj < 4; ++nj) Bm[nj] = load_fragB_rows(Bbase + (nj * 16) * 512 + ks * 32, 512, lane);
#pragma unroll
      for (int mi = 0; mi < 4; ++mi)
#pragma unroll
        for (int nj = 0; nj < 4; ++nj)
          acc[mi][nj] = wmma_f16(A[mi], Bm[nj], acc[mi][nj]);
    }
  }
  // --- epilogue 2: + proj_b, stage in LDS, coalesced f32 stores ---
  __shared__ alignas(16) float sC[64 * 64];
#pragma unroll
  for (int nj = 0; nj < 4; ++nj) {
    float pb = proj_b[nb * 64 + nj * 16 + cl];
#pragma unroll
    for (int mi = 0; mi < 4; ++mi)
#pragma unroll
      for (int g = 0; g < 8; ++g) {
        int r = mi * 16 + g + half8;
        sC[r * 64 + nj * 16 + cl] = acc[mi][nj][g] + pb;
      }
  }
  __syncthreads();
#pragma unroll
  for (int rr = 0; rr < 2; ++rr) {
    int r = lane * 2 + rr;
    const f32x4* src = reinterpret_cast<const f32x4*>(sC + r * 64);
    f32x4* dst = reinterpret_cast<f32x4*>(out + (size_t)(mb * 64 + r) * 512 + nb * 64);
#pragma unroll
    for (int j = 0; j < 16; ++j) dst[j] = src[j];
  }
}

// ---------------------------------------------------------------------------
extern "C" void kernel_launch(void* const* d_in, const int* in_sizes, int n_in,
                              void* d_out, int out_size, void* d_ws, size_t ws_size,
                              hipStream_t stream) {
  const float* x       = (const float*)d_in[0];
  const float* qkv_w   = (const float*)d_in[1];
  const float* qkv_b   = (const float*)d_in[2];
  const float* proj_w  = (const float*)d_in[3];
  const float* proj_b  = (const float*)d_in[4];
  const float* rpb     = (const float*)d_in[5];
  const float* temp    = (const float*)d_in[6];
  const float* qkvC_w  = (const float*)d_in[7];
  const float* dwC_w   = (const float*)d_in[8];
  const float* projC_w = (const float*)d_in[9];
  const float* projC_b = (const float*)d_in[10];
  const float* gao1_w  = (const float*)d_in[11];
  const float* gao1_b  = (const float*)d_in[12];
  const float* bn_g    = (const float*)d_in[13];
  const float* bn_b    = (const float*)d_in[14];
  const float* gao2_w  = (const float*)d_in[15];
  const float* gao2_b  = (const float*)d_in[16];
  float* out = (float*)d_out;

  // Workspace layout (bytes); aliases are safe due to stream-ordered launches:
  //   x_h   [67 MB]  -> reused as x1pre after gemm_qkv consumed it
  //   qkv_out[403 MB] -> head reused as x2pre after spatial_attn/dwconv done
  char* ws = (char*)d_ws;
  f16* x_h     = (f16*)(ws);                          // 67,108,864
  f16* x1pre   = x_h;
  f16* qkv_out = (f16*)(ws + (size_t)67108864);       // 402,653,184
  f16* x2pre   = qkv_out;
  f16* dwout   = (f16*)(ws + (size_t)469762048);      // 201,326,592
  f16* wcat    = (f16*)(ws + (size_t)671088640);      // 3,145,728
  f16* wp_h    = (f16*)(ws + (size_t)674234368);      // 524,288
  f16* wc_h    = (f16*)(ws + (size_t)674758656);      // 524,288
  float* bcat  = (float*)(ws + (size_t)675282944);    // 12,288
  float* gate  = (float*)(ws + (size_t)675295232);    // 256

  cvt_kernel<<<4096, 256, 0, stream>>>(x, x_h, 33554432);
  cvt_kernel<<<1536, 256, 0, stream>>>(qkv_w, wcat, 786432);
  cvt_kernel<<<1536, 256, 0, stream>>>(qkvC_w, wcat + 786432, 786432);
  cvt_kernel<<<512, 256, 0, stream>>>(proj_w, wp_h, 262144);
  cvt_kernel<<<512, 256, 0, stream>>>(projC_w, wc_h, 262144);
  bcat_kernel<<<12, 256, 0, stream>>>(qkv_b, bcat);

  gemm_qkv_kernel<<<dim3(48, 1024), 32, 0, stream>>>(x_h, wcat, bcat, qkv_out);
  dwconv_kernel<<<65536, 256, 0, stream>>>(qkv_out, dwC_w, dwout);
  spatial_attn_kernel<<<16384, 32, 0, stream>>>(qkv_out, rpb, x1pre);
  channel_attn_kernel<<<16384, 32, 0, stream>>>(dwout, temp, x2pre);
  gate_kernel<<<1, 64, 0, stream>>>(gao1_w, gao1_b, bn_g, bn_b, gao2_w, gao2_b, gate);
  final_gemm_kernel<<<dim3(8, 1024), 32, 0, stream>>>(x1pre, x2pre, wp_h, wc_h,
                                                      proj_b, projC_b, gate, out);
}
